// SymmetrizationMLP_68101001446045
// MI455X (gfx1250) — compile-verified, matrix operand order checked by reading, not run on hardware
//
#include <hip/hip_runtime.h>

typedef __attribute__((ext_vector_type(8)))  _Float16 v8h;
typedef __attribute__((ext_vector_type(16))) _Float16 v16h;
typedef __attribute__((ext_vector_type(8)))  float    v8f;

// Wave-local LDS ordering: DS ops from one wave complete in order, so waiting
// DScnt==0 makes prior per-wave LDS stores visible to this wave's later loads
// without a (cross-wave) s_barrier.
#define LDS_FENCE() asm volatile("s_wait_dscnt 0x0" ::: "memory")

#define B_SZ 16
#define N_PTS 1024
#define K_PTS 8
#define NCLS 40
#define NCLS_PAD 48
#define G_PERMS 40320
#define DIN 24
#define DIN_PAD 32
#define H1 256
#define H2 128

#define WAVES 4
#define THREADS (WAVES * 32)
#define SUB 16                                   // WMMA M per sub-tile
#define ROWS_PER_WAVE 32                         // two sub-tiles share B frags
#define PERMS_PER_ITER (WAVES * ROWS_PER_WAVE)   // 128
#define ITERS 5
#define PERMS_PER_BLOCK (PERMS_PER_ITER * ITERS) // 640
#define BLOCKS_PER_BATCH (G_PERMS / PERMS_PER_BLOCK) // 63

struct __align__(16) Smem {
  _Float16 W1t[H1 * DIN_PAD];                  // [h1][k] k-major, 16 KB
  _Float16 W2t[H2 * H1];                       // [h2][h1], 64 KB
  _Float16 W3t[NCLS_PAD * H2];                 // [c][h2], 12 KB (cols >=40 zero)
  float    b1v[H1];
  float    b2v[H2];
  float    pts[32];                            // 24 used
  _Float16 hTile[WAVES][SUB * H1];             // 16x256 per wave (reused per sub-tile)
  _Float16 h2Tile[WAVES][ROWS_PER_WAVE * H2];  // 32x128 per wave; head doubles as A-tile
};

__device__ inline v16h ld_frag(const _Float16* p0) {
  v8h lo = *(const v8h*)(p0);
  v8h hi = *(const v8h*)(p0 + 16);
  v16h r;
#pragma unroll
  for (int i = 0; i < 8; ++i) { r[i] = lo[i]; r[i + 8] = hi[i]; }
  return r;
}

// GEMM1 for one 16-row sub-tile: [16x32] x [32x256] -> hT (bias+relu, f16)
__device__ inline void gemm1_tile(const Smem& s, v16h a1, _Float16* hT,
                                  int col, int halfSel, int kb) {
#pragma unroll 1
  for (int nt = 0; nt < H1 / 16; ++nt) {
    v16h bf = ld_frag(&s.W1t[(nt * 16 + col) * DIN_PAD + kb]);
    v8f acc = {};
    acc = __builtin_amdgcn_wmma_f32_16x16x32_f16(false, a1, false, bf,
                                                 (short)0, acc, false, false);
    float bias = s.b1v[nt * 16 + col];
#pragma unroll
    for (int r = 0; r < 8; ++r) {
      int row = r + halfSel * 8;
      float v = acc[r] + bias;
      v = v > 0.0f ? v : 0.0f;
      hT[row * H1 + nt * 16 + col] = (_Float16)v;
    }
  }
}

__global__ void init_out_kernel(const float* __restrict__ b3, float* __restrict__ out) {
  int i = blockIdx.x * blockDim.x + threadIdx.x;
  if (i < B_SZ * NCLS) out[i] = b3[i % NCLS];
}

__global__ __launch_bounds__(THREADS)
void symmlp_kernel(const float* __restrict__ x, const int* __restrict__ sample_idx,
                   const int* __restrict__ perms,
                   const float* __restrict__ W1, const float* __restrict__ b1,
                   const float* __restrict__ W2, const float* __restrict__ b2,
                   const float* __restrict__ W3, const float* __restrict__ b3,
                   float* __restrict__ out) {
  __shared__ Smem s;
  const int tid  = threadIdx.x;
  const int lane = tid & 31;
  const int wave = tid >> 5;
  const int b    = blockIdx.x / BLOCKS_PER_BATCH;
  const int blk  = blockIdx.x % BLOCKS_PER_BATCH;

  // ---- stage weights into LDS (f32 -> f16, transposed to K-major) ----
  for (int i = tid; i < H1 * DIN_PAD; i += THREADS) {
    int h = i >> 5, k = i & 31;
    s.W1t[i] = (k < DIN) ? (_Float16)W1[k * H1 + h] : (_Float16)0.0f;
  }
  for (int i = tid; i < H2 * H1; i += THREADS) {
    int h2 = i >> 8, h1 = i & 255;
    s.W2t[i] = (_Float16)W2[h1 * H2 + h2];
  }
  for (int i = tid; i < NCLS_PAD * H2; i += THREADS) {
    int c = i >> 7, h = i & 127;
    s.W3t[i] = (c < NCLS) ? (_Float16)W3[h * NCLS + c] : (_Float16)0.0f;
  }
  for (int i = tid; i < H1; i += THREADS) s.b1v[i] = b1[i];
  for (int i = tid; i < H2; i += THREADS) s.b2v[i] = b2[i];
  if (tid < DIN) {
    int k = tid / 3, c = tid % 3;
    int idx = sample_idx[b * K_PTS + k];
    s.pts[tid] = x[(b * N_PTS + idx) * 3 + c];
  }
  __syncthreads();   // only cross-wave sync needed: shared weight staging

  const int col     = lane & 15;
  const int halfSel = lane >> 4;
  const int kb      = halfSel * 8;
  const float invG  = 1.0f / (float)G_PERMS;

  _Float16* hT  = s.hTile[wave];
  _Float16* h2T = s.h2Tile[wave];
  _Float16* aT  = s.h2Tile[wave];   // A-tile aliases head of h2 tile

  for (int it = 0; it < ITERS; ++it) {
    const int g0 = blk * PERMS_PER_BLOCK + it * PERMS_PER_ITER + wave * ROWS_PER_WAVE;

    // ---- gather permuted coords: 32 rows x 32 K (f16, zero-padded) ----
    for (int r = 0; r < ROWS_PER_WAVE; ++r) {
      float v = 0.0f;
      if (lane < DIN) {
        int k = lane / 3, c = lane % 3;
        int p = perms[(g0 + r) * K_PTS + k];
        v = s.pts[p * 3 + c];
      }
      aT[r * DIN_PAD + lane] = (_Float16)v;
    }
    LDS_FENCE();

    // ---- GEMM1 + A-fragment capture, sub-tile 0 then sub-tile 1 ----
    v16h a2A[H1 / 32], a2B[H1 / 32];
    {
      v16h a1 = ld_frag(&aT[(lane & 15) * DIN_PAD + kb]);
      gemm1_tile(s, a1, hT, col, halfSel, kb);
      LDS_FENCE();
#pragma unroll
      for (int ks = 0; ks < H1 / 32; ++ks)
        a2A[ks] = ld_frag(&hT[(lane & 15) * H1 + ks * 32 + kb]);
    }
    {
      v16h a1 = ld_frag(&aT[(SUB + (lane & 15)) * DIN_PAD + kb]);
      gemm1_tile(s, a1, hT, col, halfSel, kb);   // WAR on hT safe: DS in-order
      LDS_FENCE();
#pragma unroll
      for (int ks = 0; ks < H1 / 32; ++ks)
        a2B[ks] = ld_frag(&hT[(lane & 15) * H1 + ks * 32 + kb]);
    }

    // ---- GEMM2: [32x256] x [256x128] -> h2T, B fragments shared 2x ----
#pragma unroll 1
    for (int nt = 0; nt < H2 / 16; ++nt) {
      v8f acc0 = {}, acc1 = {};
#pragma unroll
      for (int ks = 0; ks < H1 / 32; ++ks) {
        v16h bf = ld_frag(&s.W2t[(nt * 16 + col) * H1 + ks * 32 + kb]);
        acc0 = __builtin_amdgcn_wmma_f32_16x16x32_f16(false, a2A[ks], false, bf,
                                                      (short)0, acc0, false, false);
        acc1 = __builtin_amdgcn_wmma_f32_16x16x32_f16(false, a2B[ks], false, bf,
                                                      (short)0, acc1, false, false);
      }
      float bias = s.b2v[nt * 16 + col];
#pragma unroll
      for (int r = 0; r < 8; ++r) {
        int row = r + halfSel * 8;
        float v0 = acc0[r] + bias; v0 = v0 > 0.0f ? v0 : 0.0f;
        float v1 = acc1[r] + bias; v1 = v1 > 0.0f ? v1 : 0.0f;
        h2T[row * H2 + nt * 16 + col]         = (_Float16)v0;
        h2T[(SUB + row) * H2 + nt * 16 + col] = (_Float16)v1;
      }
    }
    LDS_FENCE();

    // ---- GEMM3: [32x128] x [128x48] -> row-reduced, atomic mean ----
    v16h a3A[H2 / 32], a3B[H2 / 32];
#pragma unroll
    for (int ks = 0; ks < H2 / 32; ++ks) {
      a3A[ks] = ld_frag(&h2T[(lane & 15) * H2 + ks * 32 + kb]);
      a3B[ks] = ld_frag(&h2T[(SUB + (lane & 15)) * H2 + ks * 32 + kb]);
    }
#pragma unroll 1
    for (int nt = 0; nt < NCLS_PAD / 16; ++nt) {
      v8f acc0 = {}, acc1 = {};
#pragma unroll
      for (int ks = 0; ks < H2 / 32; ++ks) {
        v16h bf = ld_frag(&s.W3t[(nt * 16 + col) * H2 + ks * 32 + kb]);
        acc0 = __builtin_amdgcn_wmma_f32_16x16x32_f16(false, a3A[ks], false, bf,
                                                      (short)0, acc0, false, false);
        acc1 = __builtin_amdgcn_wmma_f32_16x16x32_f16(false, a3B[ks], false, bf,
                                                      (short)0, acc1, false, false);
      }
      float ssum = 0.0f;
#pragma unroll
      for (int r = 0; r < 8; ++r) ssum += acc0[r] + acc1[r];
      ssum += __shfl_xor(ssum, 16, 32);   // fold upper row-half onto lower
      int c = nt * 16 + col;
      if (halfSel == 0 && c < NCLS)
        atomicAdd(&out[b * NCLS + c], ssum * invG);
    }
    LDS_FENCE();   // orders a3 loads before next iteration's A-tile stores
  }
  (void)b3;
}

extern "C" void kernel_launch(void* const* d_in, const int* in_sizes, int n_in,
                              void* d_out, int out_size, void* d_ws, size_t ws_size,
                              hipStream_t stream) {
  const float* x          = (const float*)d_in[0];
  const int*   sample_idx = (const int*)d_in[1];
  const int*   perms      = (const int*)d_in[2];
  const float* W1         = (const float*)d_in[3];
  const float* b1         = (const float*)d_in[4];
  const float* W2         = (const float*)d_in[5];
  const float* b2         = (const float*)d_in[6];
  const float* W3         = (const float*)d_in[7];
  const float* b3         = (const float*)d_in[8];
  float* out = (float*)d_out;

  init_out_kernel<<<(B_SZ * NCLS + 255) / 256, 256, 0, stream>>>(b3, out);
  symmlp_kernel<<<B_SZ * BLOCKS_PER_BATCH, THREADS, 0, stream>>>(
      x, sample_idx, perms, W1, b1, W2, b2, W3, b3, out);
  (void)in_sizes; (void)n_in; (void)out_size; (void)d_ws; (void)ws_size;
}
